// EarthAttention3D_4844723109907
// MI455X (gfx1250) — matrix operand order repaired and checked by reading, not compiled, not run on hardware
//
#include <hip/hip_runtime.h>
#include <hip/hip_bf16.h>

typedef __bf16 bf16_t;
typedef __bf16 v16bf __attribute__((ext_vector_type(16)));
typedef float  v8f   __attribute__((ext_vector_type(8)));

union AFrag { v16bf v; unsigned u[8]; bf16_t e[16]; };

#define NTOK 144
#define DIMC 192
#define HD   32
#define NH   6
#define NWIN 32
#define XS   200   // x / out LDS stride (bf16 elems)
#define QS   40    // q,k LDS stride (bf16 elems)
#define VTS  168   // vT LDS stride (bf16 elems), padded past 160
#define SS   164   // scores LDS stride (f32 elems), padded past 160

// A-fragment (16x32 bf16, M x K) from row-major bf16 buffer.
// lanes 0-15: M=lane, V0-3 K=0..7, V4-7 K=16..23 ; lanes 16-31: K+8.
__device__ __forceinline__ v16bf load_afrag(const bf16_t* buf, int row0, int k0,
                                            int stride, int lane) {
  AFrag f;
  const bf16_t* p = buf + (row0 + (lane & 15)) * stride;
  const int kb = k0 + ((lane >> 4) << 3);
#pragma unroll
  for (int v = 0; v < 4; ++v) f.u[v] = *(const unsigned*)(p + kb + 2 * v);
#pragma unroll
  for (int v = 4; v < 8; ++v) f.u[v] = *(const unsigned*)(p + kb + 16 + 2 * (v - 4));
  return f.v;
}

// A-fragment with on-the-fly f32 -> bf16 conversion (for softmaxed scores).
__device__ __forceinline__ v16bf load_afrag_f32(const float* buf, int row0, int k0,
                                                int stride, int lane) {
  AFrag f;
  const float* p = buf + (row0 + (lane & 15)) * stride;
  const int kb = k0 + ((lane >> 4) << 3);
#pragma unroll
  for (int v = 0; v < 8; ++v) {
    const int k = kb + ((v < 4) ? (2 * v) : (16 + 2 * (v - 4)));
    f.e[2 * v]     = (bf16_t)p[k];
    f.e[2 * v + 1] = (bf16_t)p[k + 1];
  }
  return f.v;
}

// B-fragment (32x16 bf16, K x N) where source is row-major [n][k] (K contiguous).
// lanes 0-15: N=lane, V0-3 K=0..7, V4-7 K=16..23 ; lanes 16-31: K+8.
__device__ __forceinline__ v16bf load_bfrag(const bf16_t* buf, int n0, int k0,
                                            int stride, int lane) {
  AFrag f;
  const bf16_t* p = buf + (size_t)(n0 + (lane & 15)) * stride + k0 + ((lane >> 4) << 3);
#pragma unroll
  for (int v = 0; v < 8; ++v) f.u[v] = *(const unsigned*)(p + ((v < 4) ? (2 * v) : (16 + 2 * (v - 4))));
  return f.v;
}

__device__ __forceinline__ v8f wmma_bf16(v16bf a, v16bf b, v8f c) {
  return __builtin_amdgcn_wmma_f32_16x16x32_bf16(false, a, false, b, (short)0, c,
                                                 false, false);
}

// ---------------------------------------------------------------------------
// Kernel 1: one-time f32 -> bf16 conversion of qkv_w (576x192) and proj_w
// (192x192) into workspace (stays resident in L2 across all 480 blocks).
// ---------------------------------------------------------------------------
__global__ void convert_weights_kernel(const float* __restrict__ qkvw,
                                       const float* __restrict__ projw,
                                       bf16_t* __restrict__ qkvw_bf,
                                       bf16_t* __restrict__ projw_bf) {
  const int i = blockIdx.x * blockDim.x + threadIdx.x;
  if (i < 3 * DIMC * DIMC) qkvw_bf[i] = (bf16_t)qkvw[i];
  if (i < DIMC * DIMC)     projw_bf[i] = (bf16_t)projw[i];
}

// ---------------------------------------------------------------------------
// Kernel 2: fused window attention + projection. One block per (b, w) window.
// ---------------------------------------------------------------------------
__global__ __launch_bounds__(256, 1) void earth_attn_kernel(
    const float* __restrict__ x, const float* __restrict__ mask,
    const bf16_t* __restrict__ qkvw, const float* __restrict__ qkvb,
    const float* __restrict__ bias_table, const int* __restrict__ posidx,
    const bf16_t* __restrict__ projw, const float* __restrict__ projb,
    float* __restrict__ out) {
  __shared__ bf16_t sX[NTOK * XS];    // x window, bf16
  __shared__ bf16_t sQ[NTOK * QS];    // q_h  [tok][feat]
  __shared__ bf16_t sK[NTOK * QS];    // k_h  [tok][feat]
  __shared__ bf16_t sVT[HD * VTS];    // v_hᵀ [feat][tok], zero-padded K
  __shared__ float  sS[NTOK * SS];    // scores f32, zero-padded K
  __shared__ bf16_t sO[NTOK * XS];    // attn@v result, all heads

  const int tid  = threadIdx.x;
  const int lane = tid & 31;
  const int wid  = tid >> 5;
  const int w    = blockIdx.x % NWIN;     // window-type index (for bias table)
  const float scale = 0.17677669529663689f;  // 32^-0.5

  const float* xg    = x    + (size_t)blockIdx.x * NTOK * DIMC;
  const float* maskw = mask + (size_t)blockIdx.x * NTOK * NTOK;

  // Stage x -> bf16 LDS; zero the K-padding of scores and vT once.
  for (int i = tid; i < NTOK * DIMC; i += 256) {
    sX[(i / DIMC) * XS + (i % DIMC)] = (bf16_t)xg[i];
  }
  for (int i = tid; i < NTOK * (SS - NTOK); i += 256) {
    sS[(i / (SS - NTOK)) * SS + NTOK + (i % (SS - NTOK))] = 0.0f;
  }
  for (int i = tid; i < HD * (VTS - NTOK); i += 256) {
    sVT[(i / (VTS - NTOK)) * VTS + NTOK + (i % (VTS - NTOK))] = (bf16_t)0.0f;
  }
  __syncthreads();

  for (int h = 0; h < NH; ++h) {
    // ---- Step 1: q_h, k_h, v_hT = x @ Wqkv[:, head cols] (+bias, q scaled) ----
    // 9 m-tiles x 6 n-tiles (2 q, 2 k, 2 v), K=192 in 6 steps.
    for (int t = wid; t < 54; t += 8) {
      const int mt  = t / 6, nt = t % 6;
      const int sec = nt >> 1;              // 0=q 1=k 2=v
      const int nl  = (nt & 1) << 4;        // 0 or 16 within head cols
      const int drow = sec * DIMC + h * HD + nl;  // row base in qkv_w
      v8f acc = {};
#pragma unroll
      for (int kk = 0; kk < 6; ++kk) {
        v16bf a = load_afrag(sX, mt * 16, kk * 32, XS, lane);
        v16bf b = load_bfrag(qkvw, drow, kk * 32, DIMC, lane);
        acc = wmma_bf16(a, b, acc);
      }
      const int   ncol = nl + (lane & 15);
      const float bv   = qkvb[drow + (lane & 15)];
      const int   mrow = mt * 16 + ((lane >> 4) << 3);
      if (sec == 0) {
#pragma unroll
        for (int r = 0; r < 8; ++r)
          sQ[(mrow + r) * QS + ncol] = (bf16_t)((acc[r] + bv) * scale);
      } else if (sec == 1) {
#pragma unroll
        for (int r = 0; r < 8; ++r)
          sK[(mrow + r) * QS + ncol] = (bf16_t)(acc[r] + bv);
      } else {
#pragma unroll
        for (int r = 0; r < 8; ++r)
          sVT[ncol * VTS + (mrow + r)] = (bf16_t)(acc[r] + bv);
      }
    }
    __syncthreads();

    // ---- Step 2: S = q kᵀ + bias_gather + mask (9x9 tiles, single K=32) ----
    for (int t = wid; t < 81; t += 8) {
      const int it = t / 9, jt = t % 9;
      v16bf a = load_afrag(sQ, it * 16, 0, QS, lane);
      v16bf b = load_bfrag(sK, jt * 16, 0, QS, lane);
      v8f acc = {};
      acc = wmma_bf16(a, b, acc);
      const int i0 = it * 16 + ((lane >> 4) << 3);
      const int j  = jt * 16 + (lane & 15);
#pragma unroll
      for (int r = 0; r < 8; ++r) {
        const int i  = i0 + r;
        const int pi = posidx[i * NTOK + j];
        const float bias = bias_table[(size_t)pi * (NWIN * NH) + w * NH + h];
        sS[i * SS + j] = acc[r] + bias + maskw[i * NTOK + j];
      }
    }
    __syncthreads();

    // ---- Step 3: row softmax, one wave32 per row ----
    for (int row = wid; row < NTOK; row += 8) {
      float* srow = sS + row * SS;
      float m = -1e30f;
      for (int c = lane; c < NTOK; c += 32) m = fmaxf(m, srow[c]);
#pragma unroll
      for (int off = 16; off > 0; off >>= 1) m = fmaxf(m, __shfl_xor(m, off, 32));
      float sum = 0.0f;
      for (int c = lane; c < NTOK; c += 32) {
        const float e = __expf(srow[c] - m);
        srow[c] = e;
        sum += e;
      }
#pragma unroll
      for (int off = 16; off > 0; off >>= 1) sum += __shfl_xor(sum, off, 32);
      const float inv = 1.0f / sum;
      for (int c = lane; c < NTOK; c += 32) srow[c] *= inv;
    }
    __syncthreads();

    // ---- Step 4: O_h = P @ v_h  (9x2 tiles, K padded 144 -> 160) ----
    for (int t = wid; t < 18; t += 8) {
      const int mt = t / 2, nt = t % 2;
      v8f acc = {};
#pragma unroll
      for (int kk = 0; kk < 5; ++kk) {
        v16bf a = load_afrag_f32(sS, mt * 16, kk * 32, SS, lane);
        v16bf b = load_bfrag(sVT, nt * 16, kk * 32, VTS, lane);
        acc = wmma_bf16(a, b, acc);
      }
      const int mrow = mt * 16 + ((lane >> 4) << 3);
      const int col  = h * HD + nt * 16 + (lane & 15);
#pragma unroll
      for (int r = 0; r < 8; ++r) sO[(mrow + r) * XS + col] = (bf16_t)acc[r];
    }
    __syncthreads();  // protect sQ/sK/sVT/sS before next head
  }

  // ---- Step 5: out = O @ Wprojᵀ + b  (9x12 tiles, K=192 in 6 steps) ----
  float* outg = out + (size_t)blockIdx.x * NTOK * DIMC;
  for (int t = wid; t < 108; t += 8) {
    const int mt = t / 12, nt = t % 12;
    v8f acc = {};
#pragma unroll
    for (int kk = 0; kk < 6; ++kk) {
      v16bf a = load_afrag(sO, mt * 16, kk * 32, XS, lane);
      v16bf b = load_bfrag(projw, nt * 16, kk * 32, DIMC, lane);
      acc = wmma_bf16(a, b, acc);
    }
    const int col  = nt * 16 + (lane & 15);
    const float pb = projb[col];
    const int mrow = mt * 16 + ((lane >> 4) << 3);
#pragma unroll
    for (int r = 0; r < 8; ++r)
      outg[(size_t)(mrow + r) * DIMC + col] = acc[r] + pb;
  }
}

extern "C" void kernel_launch(void* const* d_in, const int* in_sizes, int n_in,
                              void* d_out, int out_size, void* d_ws, size_t ws_size,
                              hipStream_t stream) {
  const float* x          = (const float*)d_in[0];
  const float* mask       = (const float*)d_in[1];
  const float* qkv_w      = (const float*)d_in[2];
  const float* qkv_b      = (const float*)d_in[3];
  const float* proj_w     = (const float*)d_in[4];
  const float* proj_b     = (const float*)d_in[5];
  const float* bias_table = (const float*)d_in[6];
  const int*   posidx     = (const int*)d_in[7];

  char* ws = (char*)d_ws;
  bf16_t* qkvw_bf = (bf16_t*)ws;                  // 576*192*2 = 221184 B
  bf16_t* projw_bf = (bf16_t*)(ws + 221184);      // 192*192*2 =  73728 B

  convert_weights_kernel<<<(3 * DIMC * DIMC + 255) / 256, 256, 0, stream>>>(
      qkv_w, proj_w, qkvw_bf, projw_bf);

  earth_attn_kernel<<<480, 256, 0, stream>>>(
      x, mask, qkvw_bf, qkv_b, bias_table, posidx, projw_bf, proj_b,
      (float*)d_out);
}
